// GraphAttentionConv_61160334295757
// MI455X (gfx1250) — matrix operand order repaired and checked by reading, not compiled, not run on hardware
//
#include <hip/hip_runtime.h>
#include <hip/hip_bf16.h>

// ---------------------------------------------------------------------------
// GraphAttentionConv for MI455X (gfx1250, wave32).
//
// Pipeline (fused/factored to minimize edge-stream memory traffic):
//   K0 precompute_P : P[16][4] = We^T folded with (q_even+q_odd); qb[4] from be
//   K0b convert_W   : Whalf = (f16)W  (32KB constant, converted once)
//   K1 node_gemm    : hidden = x @ W^T + b   (v_wmma_f32_16x16x32_f16)
//   K2 node_coeff   : a[n,h] = q_even . hidden,  b[n,h] = q_odd . hidden
//   K3 init_ws      : zero d_out / normsum / counts, wmax keys = 0 (== -inf)
//   K4 edge_pass1   : w = leaky_relu(a_in + b_out + ef.P + qb); atomicMax key
//   K5 edge_pass2   : att = exp(w - wmax_out) * ew; atomicAdd normsum, counts
//   K6 edge_pass3   : d_out[out,:] += att * hidden[in,:]   (L2-resident atomics)
//   K7 node_final   : out = relu(out / ((normsum/cnt + eps) * cnt))
// ---------------------------------------------------------------------------

typedef __attribute__((ext_vector_type(16))) _Float16 v16h;
typedef __attribute__((ext_vector_type(8)))  float    v8f;

#define IN_DIM   128
#define OUT_DIM  64
#define EDGE_DIM 16
#define HEADS    4
#define NEG_SLOPE 0.2f
#define EPS 1e-10f

// --- K0: fold query into We / be -------------------------------------------
__global__ void precompute_P(const float* __restrict__ We,
                             const float* __restrict__ be,
                             const float* __restrict__ query,
                             float* __restrict__ P, float* __restrict__ qb) {
  int t = threadIdx.x;
  if (t < 64) {
    int j = t & 15, h = t >> 4;
    float s = 0.f;
    for (int dl = 0; dl < 16; ++dl) {
      int d = h * 16 + dl;
      float Qe = query[h * 32 + 2 * dl] + query[h * 32 + 2 * dl + 1];
      s += We[d * EDGE_DIM + j] * Qe;
    }
    P[j * HEADS + h] = s;
  }
  if (t < HEADS) {
    float s = 0.f;
    for (int dl = 0; dl < 16; ++dl) {
      int d = t * 16 + dl;
      s += be[d] * (query[t * 32 + 2 * dl] + query[t * 32 + 2 * dl + 1]);
    }
    qb[t] = s;
  }
}

// --- K0b: one-time f32 -> f16 conversion of W (row-major 64x128) -----------
__global__ void convert_W(const float* __restrict__ W,
                          _Float16* __restrict__ Whalf, int n) {
  int t = blockIdx.x * blockDim.x + threadIdx.x;
  if (t < n) Whalf[t] = (_Float16)W[t];
}

// --- K1: hidden = x @ W^T + b via WMMA -------------------------------------
// One wave per 16-row tile of x; 4 column tiles of 16 (OUT_DIM=64); K=128 in
// 4 steps of 32. A fragment (16x32 f16): lane L holds row M=L%16, halves
// [0..7] = K(half*16 + 0..7), [8..15] = K(half*16 + 8 + 0..7), half = L/16.
// B fragment (32x16 f16): lane L holds col N=L%16, halves [i] = K(half*16+i)
// -> one contiguous 32-byte packed-f16 vector load from Whalf, no cvts.
// C/D (16x16 f32): VGPR r -> row (half*8 + r), col N=L%16.
__global__ void node_gemm(const float* __restrict__ x,
                          const _Float16* __restrict__ Whalf,
                          const float* __restrict__ bias,
                          float* __restrict__ hidden, int N) {
  int wavesPerBlock = blockDim.x >> 5;
  int wave = blockIdx.x * wavesPerBlock + (threadIdx.x >> 5);
  int lane = threadIdx.x & 31;
  int numTiles = (N + 15) >> 4;
  if (wave >= numTiles) return;              // wave-uniform: EXEC stays all-1s

  int row0 = wave << 4;
  int half = lane >> 4;
  int l15  = lane & 15;
  int mrow = row0 + l15;
  if (mrow > N - 1) mrow = N - 1;            // clamp loads (stores guarded)

  v8f acc[4];
  #pragma unroll
  for (int c = 0; c < 4; ++c)
    acc[c] = (v8f){0.f, 0.f, 0.f, 0.f, 0.f, 0.f, 0.f, 0.f};

  const float* xrow = x + (size_t)mrow * IN_DIM;
  #pragma unroll
  for (int k0 = 0; k0 < IN_DIM; k0 += 32) {
    if (k0 + 32 < IN_DIM)
      __builtin_prefetch(xrow + k0 + 32, 0, 3);   // global_prefetch_b8
    v16h afrag;
    #pragma unroll
    for (int j = 0; j < 8; ++j) {
      afrag[j]     = (_Float16)xrow[k0 + half * 16 + j];
      afrag[8 + j] = (_Float16)xrow[k0 + half * 16 + 8 + j];
    }
    #pragma unroll
    for (int c = 0; c < 4; ++c) {
      int ncol = c * 16 + l15;
      // 32-byte aligned packed-f16 fragment (two global_load_b128)
      v16h bfrag = *(const v16h*)(Whalf + (size_t)ncol * IN_DIM + k0 + half * 16);
      acc[c] = __builtin_amdgcn_wmma_f32_16x16x32_f16(
          false, afrag, false, bfrag, (short)0, acc[c], false, false);
    }
  }

  #pragma unroll
  for (int c = 0; c < 4; ++c) {
    int col = c * 16 + l15;
    float bv = bias[col];
    #pragma unroll
    for (int r = 0; r < 8; ++r) {
      int mr = row0 + half * 8 + r;
      if (mr < N) hidden[(size_t)mr * OUT_DIM + col] = acc[c][r] + bv;
    }
  }
}

// --- K2: per-node query projections ----------------------------------------
__global__ void node_coeff(const float* __restrict__ hidden,
                           const float* __restrict__ query,
                           float* __restrict__ aq, float* __restrict__ bq,
                           int N) {
  int t = blockIdx.x * blockDim.x + threadIdx.x;
  if (t >= N * HEADS) return;
  int n = t >> 2, h = t & 3;
  const float* hv = hidden + (size_t)n * OUT_DIM + h * 16;
  const float* q  = query + h * 32;
  float s0 = 0.f, s1 = 0.f;
  #pragma unroll
  for (int dl = 0; dl < 16; ++dl) {
    float hd = hv[dl];
    s0 += q[2 * dl] * hd;
    s1 += q[2 * dl + 1] * hd;
  }
  aq[t] = s0;
  bq[t] = s1;
}

// --- K3: zero accumulators --------------------------------------------------
__global__ void init_ws(float* __restrict__ out, float* __restrict__ normsum,
                        unsigned* __restrict__ wmaxkey,
                        float* __restrict__ countf, int N) {
  int t = blockIdx.x * blockDim.x + threadIdx.x;
  if (t < N * OUT_DIM) out[t] = 0.f;
  if (t < N * HEADS) { normsum[t] = 0.f; wmaxkey[t] = 0u; }
  if (t < N) countf[t] = 0.f;
}

__device__ __forceinline__ unsigned f2key(float f) {
  unsigned u = __float_as_uint(f);
  return (u & 0x80000000u) ? ~u : (u | 0x80000000u);
}
__device__ __forceinline__ float key2f(unsigned k) {
  return __uint_as_float((k & 0x80000000u) ? (k ^ 0x80000000u) : ~k);
}

// --- K4: logits + segment max ----------------------------------------------
__global__ void edge_pass1(const int* __restrict__ el,
                           const float* __restrict__ ef,
                           const float* __restrict__ aq,
                           const float* __restrict__ bq,
                           const float* __restrict__ P,
                           const float* __restrict__ qb,
                           float* __restrict__ w,
                           unsigned* __restrict__ wmaxkey, int E, int M) {
  int m = blockIdx.x * blockDim.x + threadIdx.x;
  if (m >= M) return;
  int nin, nout;
  float ec[HEADS] = {0.f, 0.f, 0.f, 0.f};
  if (m < E) {
    nin  = el[2 * m];
    nout = el[2 * m + 1];
    const float* f = ef + (size_t)m * EDGE_DIM;
    #pragma unroll
    for (int j = 0; j < EDGE_DIM; ++j) {
      float fj = f[j];
      #pragma unroll
      for (int h = 0; h < HEADS; ++h) ec[h] += fj * P[j * HEADS + h];
    }
  } else {
    nin = nout = m - E;                      // self-loop, e contribution = 0
  }
  #pragma unroll
  for (int h = 0; h < HEADS; ++h) {
    float v = aq[nin * HEADS + h] + bq[nout * HEADS + h] + ec[h] + qb[h];
    v = v > 0.f ? v : NEG_SLOPE * v;         // leaky_relu
    w[(size_t)m * HEADS + h] = v;
    atomicMax(&wmaxkey[nout * HEADS + h], f2key(v));
  }
}

// --- K5: softmax numerator + segment sums ----------------------------------
__global__ void edge_pass2(const int* __restrict__ el,
                           const float* __restrict__ ewt,
                           float* __restrict__ w,
                           const unsigned* __restrict__ wmaxkey,
                           float* __restrict__ normsum,
                           float* __restrict__ countf, int E, int M) {
  int m = blockIdx.x * blockDim.x + threadIdx.x;
  if (m >= M) return;
  int nout     = (m < E) ? el[2 * m + 1] : (m - E);
  float weight = (m < E) ? ewt[m] : 1.0f;
  #pragma unroll
  for (int h = 0; h < HEADS; ++h) {
    float wm  = key2f(wmaxkey[nout * HEADS + h]);
    float att = __expf(w[(size_t)m * HEADS + h] - wm) * weight;
    w[(size_t)m * HEADS + h] = att;          // overwrite logits with att
    atomicAdd(&normsum[nout * HEADS + h], att);
  }
  atomicAdd(&countf[nout], 1.0f);
}

// --- K6: weighted message scatter (64 threads per edge, coalesced) ----------
__global__ void edge_pass3(const int* __restrict__ el,
                           const float* __restrict__ att,
                           const float* __restrict__ hidden,
                           float* __restrict__ out, int E, int M) {
  int t = blockIdx.x * blockDim.x + threadIdx.x;
  if (t >= M * OUT_DIM) return;
  int m = t >> 6, d = t & 63;
  int nin, nout;
  if (m < E) { nin = el[2 * m]; nout = el[2 * m + 1]; }
  else       { nin = nout = m - E; }
  float a = att[(size_t)m * HEADS + (d >> 4)];
  atomicAdd(&out[(size_t)nout * OUT_DIM + d],
            a * hidden[(size_t)nin * OUT_DIM + d]);
}

// --- K7: normalize + relu ---------------------------------------------------
__global__ void node_final(float* __restrict__ out,
                           const float* __restrict__ normsum,
                           const float* __restrict__ countf, int N) {
  int t = blockIdx.x * blockDim.x + threadIdx.x;
  if (t >= N * OUT_DIM) return;
  int n = t >> 6, h = (t >> 4) & 3;
  float cnt  = countf[n];                    // >= 1 (self-loops)
  float norm = normsum[n * HEADS + h] / cnt;
  float v = out[t] / ((norm + EPS) * cnt);
  out[t] = v > 0.f ? v : 0.f;
}

// ---------------------------------------------------------------------------
extern "C" void kernel_launch(void* const* d_in, const int* in_sizes, int n_in,
                              void* d_out, int out_size, void* d_ws, size_t ws_size,
                              hipStream_t stream) {
  const int*   edge_list   = (const int*)  d_in[0];  // (E,2) int32
  const float* edge_weight = (const float*)d_in[1];  // (E,)
  const float* edge_feat   = (const float*)d_in[2];  // (E,16)
  /* d_in[3] = num_node scalar (derived from sizes instead) */
  const float* x     = (const float*)d_in[4];        // (N,128)
  const float* W     = (const float*)d_in[5];        // (64,128)
  const float* b     = (const float*)d_in[6];        // (64,)
  const float* We    = (const float*)d_in[7];        // (64,16)
  const float* be    = (const float*)d_in[8];        // (64,)
  const float* query = (const float*)d_in[9];        // (4,32)
  float* out = (float*)d_out;

  const int E = in_sizes[1];
  const int N = in_sizes[4] / IN_DIM;
  const int M = E + N;

  // Workspace carve-out (256B-aligned slices)
  char* ws = (char*)d_ws;
  size_t off = 0;
  auto carve = [&](size_t bytes) {
    void* p = ws + off;
    off = (off + bytes + 255) & ~(size_t)255;
    return p;
  };
  float*     hidden  = (float*)    carve((size_t)N * OUT_DIM * 4);
  float*     aq      = (float*)    carve((size_t)N * HEADS * 4);
  float*     bq      = (float*)    carve((size_t)N * HEADS * 4);
  float*     P       = (float*)    carve(EDGE_DIM * HEADS * 4);
  float*     qb      = (float*)    carve(HEADS * 4);
  _Float16*  Whalf   = (_Float16*) carve((size_t)OUT_DIM * IN_DIM * 2);
  float*     wbuf    = (float*)    carve((size_t)M * HEADS * 4);
  unsigned*  wmaxkey = (unsigned*) carve((size_t)N * HEADS * 4);
  float*     normsum = (float*)    carve((size_t)N * HEADS * 4);
  float*     countf  = (float*)    carve((size_t)N * 4);
  (void)ws_size; (void)n_in; (void)out_size;

  const int B = 256;
  hipLaunchKernelGGL(precompute_P, dim3(1), dim3(64), 0, stream,
                     We, be, query, P, qb);

  int wElems = OUT_DIM * IN_DIM;
  hipLaunchKernelGGL(convert_W, dim3((wElems + B - 1) / B), dim3(B), 0, stream,
                     W, Whalf, wElems);

  int tiles = (N + 15) / 16;                 // one wave (32 thr) per tile
  int gemmBlocks = (tiles + (B / 32) - 1) / (B / 32);
  hipLaunchKernelGGL(node_gemm, dim3(gemmBlocks), dim3(B), 0, stream,
                     x, Whalf, b, hidden, N);

  hipLaunchKernelGGL(node_coeff, dim3((N * HEADS + B - 1) / B), dim3(B), 0,
                     stream, hidden, query, aq, bq, N);

  hipLaunchKernelGGL(init_ws, dim3((N * OUT_DIM + B - 1) / B), dim3(B), 0,
                     stream, out, normsum, wmaxkey, countf, N);

  hipLaunchKernelGGL(edge_pass1, dim3((M + B - 1) / B), dim3(B), 0, stream,
                     edge_list, edge_feat, aq, bq, P, qb, wbuf, wmaxkey, E, M);

  hipLaunchKernelGGL(edge_pass2, dim3((M + B - 1) / B), dim3(B), 0, stream,
                     edge_list, edge_weight, wbuf, wmaxkey, normsum, countf,
                     E, M);

  long long msgThreads = (long long)M * OUT_DIM;
  hipLaunchKernelGGL(edge_pass3, dim3((unsigned)((msgThreads + B - 1) / B)),
                     dim3(B), 0, stream, edge_list, wbuf, hidden, out, E, M);

  hipLaunchKernelGGL(node_final, dim3((N * OUT_DIM + B - 1) / B), dim3(B), 0,
                     stream, out, normsum, countf, N);
}
